// MixtralMoE_2886218023293
// MI455X (gfx1250) — compile-verified, matrix operand order checked by reading, not run on hardware
//
#include <hip/hip_runtime.h>

// ---------------- problem constants (from reference setup_inputs) ----------
#define T_TOK 4096   // B*S = 2*2048 tokens
#define HDIM  1024
#define IDIM  3584
#define NEXP  8
// ---------------- tiling ---------------------------------------------------
#define KT    32     // WMMA bf16 K per step
#define KTP   40     // padded LDS K stride (keeps 16B alignment, skews banks)
#define BM    32     // tokens per block tile (2 x 16-row wmma tiles)
#define BN    128    // columns per block tile (8 x 16-col wmma tiles)
#define NST1  (HDIM / KT)   // 32 k-steps in ffn1
#define NST2  (IDIM / KT)   // 112 k-steps in ffn2

typedef __attribute__((ext_vector_type(16))) __bf16 bf16x16;
typedef __attribute__((ext_vector_type(8)))  float  f32x8;

struct __align__(16) U4 { unsigned int x, y, z, w; };
union FragBF { U4 c[2]; bf16x16 v; };

__device__ __forceinline__ unsigned short f2bf(float f) {
  unsigned int u = __float_as_uint(f);
  u += 0x7FFFu + ((u >> 16) & 1u);          // round-to-nearest-even
  return (unsigned short)(u >> 16);
}

// ---- CDNA5 async global->LDS staging (ASYNCcnt-tracked DMA, no VGPR data) --
__device__ __forceinline__ unsigned lds_off32(const void* p) {
  return (unsigned)(unsigned long long)(uintptr_t)p;   // low 32b = LDS offset
}
__device__ __forceinline__ void async_ld_b128(unsigned lds_off, const void* g) {
  asm volatile("global_load_async_to_lds_b128 %0, %1, off"
               :: "v"(lds_off), "v"(g) : "memory");
}
__device__ __forceinline__ void async_ld_b64(unsigned lds_off, const void* g) {
  asm volatile("global_load_async_to_lds_b64 %0, %1, off"
               :: "v"(lds_off), "v"(g) : "memory");
}
// async loads retire in order: wait until only the newest N remain in flight
#define WAIT_ASYNC_LE(n) asm volatile("s_wait_asynccnt " #n ::: "memory")

// Fragment loader for 16-bit A/B operands of v_wmma_f32_16x16x32_bf16.
// LDS layout: row-major [row][K] with KTP stride. Per ISA 7.12.2:
// lane L -> row = base + (L&15); chunks: grp=L>>4 -> K=[grp*8..+8) and
// K=[16+grp*8..+8). Each chunk = 16 bytes -> ds_load_b128.
__device__ __forceinline__ FragBF load_frag(const unsigned short* lds,
                                            int rowbase, int lane) {
  FragBF f;
  const unsigned short* p = lds + (rowbase + (lane & 15)) * KTP + (lane >> 4) * 8;
  f.c[0] = *(const U4*)p;
  f.c[1] = *(const U4*)(p + 16);
  return f;
}

#define WMMA_BF16(A, B, C) \
  __builtin_amdgcn_wmma_f32_16x16x32_bf16(false, (A), false, (B), (short)0, (C), false, false)

// ---------------------------------------------------------------------------
// Kernel 0: zero out + counters, convert x f32 -> bf16
// ---------------------------------------------------------------------------
__global__ __launch_bounds__(256) void moe_prep(const float* __restrict__ x,
                                                float* __restrict__ out,
                                                unsigned short* __restrict__ xb,
                                                int* __restrict__ counts) {
  const long long n = (long long)T_TOK * HDIM;
  const long long gid = (long long)blockIdx.x * blockDim.x + threadIdx.x;
  if (gid < NEXP) counts[gid] = 0;
  for (long long i = gid; i < n; i += (long long)gridDim.x * blockDim.x) {
    out[i] = 0.0f;
    xb[i]  = f2bf(x[i]);
  }
}

// ---------------------------------------------------------------------------
// Kernel 0b: per-expert weight convert+transpose  [R][C] f32 -> [C][R] bf16
// 352MB read + 176MB write ~= 23us at 23.3 TB/s; bf16 weights fit in 192MB L2
// ---------------------------------------------------------------------------
__global__ __launch_bounds__(256) void moe_tcvt(const float* __restrict__ src,
                                                unsigned short* __restrict__ dst,
                                                int R, int C) {
  const int e = blockIdx.z;
  src += (size_t)e * R * C;
  dst += (size_t)e * R * C;
  const int r0 = blockIdx.y * 32, c0 = blockIdx.x * 32;
  __shared__ unsigned short lt[32][33];
  const int tr  = threadIdx.x >> 3;        // 0..31
  const int tc4 = (threadIdx.x & 7) * 4;   // 0..28
  const float4 f = *(const float4*)&src[(size_t)(r0 + tr) * C + c0 + tc4];
  lt[tc4 + 0][tr] = f2bf(f.x);
  lt[tc4 + 1][tr] = f2bf(f.y);
  lt[tc4 + 2][tr] = f2bf(f.z);
  lt[tc4 + 3][tr] = f2bf(f.w);
  __syncthreads();
  const unsigned int lo = (unsigned)lt[tr][tc4] | ((unsigned)lt[tr][tc4 + 1] << 16);
  const unsigned int hi = (unsigned)lt[tr][tc4 + 2] | ((unsigned)lt[tr][tc4 + 3] << 16);
  uint2 o; o.x = lo; o.y = hi;
  *(uint2*)&dst[(size_t)(c0 + tr) * R + r0 + tc4] = o;
}

// ---------------------------------------------------------------------------
// Kernel 1: router — logits, softmax, top-2, renormalize, build expert lists
// ---------------------------------------------------------------------------
__global__ __launch_bounds__(256) void moe_router(const float* __restrict__ x,
                                                  const float* __restrict__ gw,
                                                  int* __restrict__ counts,
                                                  int* __restrict__ list_tok,
                                                  float* __restrict__ list_w) {
  const int t = blockIdx.x * blockDim.x + threadIdx.x;
  if (t >= T_TOK) return;
  const float* xr = x + (long long)t * HDIM;
  float acc[NEXP];
#pragma unroll
  for (int e = 0; e < NEXP; ++e) acc[e] = 0.0f;
  for (int h = 0; h < HDIM; ++h) {
    const float xv = xr[h];
#pragma unroll
    for (int e = 0; e < NEXP; ++e) acc[e] += xv * gw[e * HDIM + h];
  }
  float mx = acc[0];
#pragma unroll
  for (int e = 1; e < NEXP; ++e) mx = fmaxf(mx, acc[e]);
  float p[NEXP];
#pragma unroll
  for (int e = 0; e < NEXP; ++e) p[e] = __expf(acc[e] - mx);
  int i0 = 0;
#pragma unroll
  for (int e = 1; e < NEXP; ++e) if (p[e] > p[i0]) i0 = e;
  int i1 = (i0 == 0) ? 1 : 0;
#pragma unroll
  for (int e = 0; e < NEXP; ++e) if (e != i0 && p[e] > p[i1]) i1 = e;
  const float denom = p[i0] + p[i1];
  const float w0 = p[i0] / denom, w1 = p[i1] / denom;

  int s0 = atomicAdd(&counts[i0], 1);
  list_tok[i0 * T_TOK + s0] = t;  list_w[i0 * T_TOK + s0] = w0;
  int s1 = atomicAdd(&counts[i1], 1);
  list_tok[i1 * T_TOK + s1] = t;  list_w[i1 * T_TOK + s1] = w1;
}

// ---------------------------------------------------------------------------
// Kernel 2: exclusive prefix over 8 expert counts -> compact h-row offsets
// ---------------------------------------------------------------------------
__global__ void moe_prefix(const int* __restrict__ counts,
                           int* __restrict__ offsets) {
  if (threadIdx.x == 0) {
    int s = 0;
    for (int e = 0; e < NEXP; ++e) { offsets[e] = s; s += counts[e]; }
  }
}

// ---------------------------------------------------------------------------
// Kernel 3: grouped GEMM1  h = silu(x*w1) ⊙ (x*w3)   [per-expert token rows]
// Double-buffered async-DMA pipeline: stage k+1 overlaps WMMA on stage k.
// grid: (IDIM/BN, T/BM, NEXP), block 256 (8 wave32)
// ---------------------------------------------------------------------------
__global__ __launch_bounds__(256) void moe_ffn1(
    const unsigned short* __restrict__ xb, const unsigned short* __restrict__ w1t,
    const unsigned short* __restrict__ w3t, const int* __restrict__ list_tok,
    const int* __restrict__ counts, const int* __restrict__ offsets,
    unsigned short* __restrict__ hb) {
  const int e   = blockIdx.z;
  const int mt  = blockIdx.y;
  const int n0  = blockIdx.x * BN;
  const int cnt = counts[e];
  if (mt * BM >= cnt) return;                  // block-uniform early exit
  const int offE = offsets[e];

  __shared__ __align__(16) unsigned short la [2][BM * KTP];
  __shared__ __align__(16) unsigned short lb1[2][BN * KTP];
  __shared__ __align__(16) unsigned short lb3[2][BN * KTP];

  const int tid  = threadIdx.x;
  const int lane = tid & 31;
  const int wave = tid >> 5;
  const int rowt = (wave >> 2) * 16;           // 0 or 16
  const int colb = (wave & 3) * 32;            // wave owns 2 n-tiles

  // A-tile gather coords (fixed per thread): 8B per thread
  const int arow  = tid >> 3;                  // 0..31
  const int acg   = (tid & 7) * 4;             // 4 bf16
  const int aslot = mt * BM + arow;
  const int atok  = (aslot < cnt) ? list_tok[e * T_TOK + aslot] : 0;
  const unsigned short* asrc = xb + (unsigned long long)atok * HDIM + acg;

  // B-tile coords: 2 x 16B chunks per thread per matrix
  const int br0 = tid >> 2;                    // 0..63
  const int bck = (tid & 3) * 8;               // 0,8,16,24
  const unsigned long long wtbase = (unsigned long long)e * IDIM * HDIM;
  const unsigned short* b1src = w1t + wtbase + (unsigned long long)(n0 + br0) * HDIM + bck;
  const unsigned short* b3src = w3t + wtbase + (unsigned long long)(n0 + br0) * HDIM + bck;

  unsigned aoff[2], b1off[2], b3off[2];
#pragma unroll
  for (int b = 0; b < 2; ++b) {
    aoff[b]  = lds_off32(&la [b][arow * KTP + acg]);
    b1off[b] = lds_off32(&lb1[b][br0 * KTP + bck]);
    b3off[b] = lds_off32(&lb3[b][br0 * KTP + bck]);
  }

  auto issue = [&](int it, int buf) {          // 5 async ops per wave
    const int kt = it * KT;
    async_ld_b64 (aoff[buf],                 asrc + kt);
    async_ld_b128(b1off[buf],                b1src + kt);
    async_ld_b128(b1off[buf] + 64 * KTP * 2, b1src + kt + (unsigned long long)64 * HDIM);
    async_ld_b128(b3off[buf],                b3src + kt);
    async_ld_b128(b3off[buf] + 64 * KTP * 2, b3src + kt + (unsigned long long)64 * HDIM);
  };

  f32x8 accg0 = {}, accg1 = {}, accu0 = {}, accu1 = {};

  issue(0, 0);
  for (int it = 0; it < NST1; ++it) {
    const int cur = it & 1;
    if (it + 1 < NST1) {
      issue(it + 1, cur ^ 1);
      if (it + 2 < NST1) {
        __builtin_prefetch(b1src + (it + 2) * KT, 0, 1);
        __builtin_prefetch(b3src + (it + 2) * KT, 0, 1);
      }
      WAIT_ASYNC_LE(5);                        // stage `it` landed; it+1 in flight
    } else {
      WAIT_ASYNC_LE(0);
    }
    __syncthreads();                           // publish all waves' staging

    const FragBF fa  = load_frag(la [cur], rowt,      lane);
    const FragBF b10 = load_frag(lb1[cur], colb,      lane);
    const FragBF b30 = load_frag(lb3[cur], colb,      lane);
    const FragBF b11 = load_frag(lb1[cur], colb + 16, lane);
    const FragBF b31 = load_frag(lb3[cur], colb + 16, lane);
    accg0 = WMMA_BF16(fa.v, b10.v, accg0);
    accu0 = WMMA_BF16(fa.v, b30.v, accu0);
    accg1 = WMMA_BF16(fa.v, b11.v, accg1);
    accu1 = WMMA_BF16(fa.v, b31.v, accu1);
    __syncthreads();                           // done reading `cur` buffer
  }

  // ---- epilogue: h = silu(g)*u, store bf16 at compact slot rows ----
  const int nl = lane & 15;
  const int mh = (lane >> 4) * 8;
#pragma unroll
  for (int i = 0; i < 8; ++i) {
    const int sl = mt * BM + rowt + mh + i;
    if (sl < cnt) {
      const unsigned long long hrow = (unsigned long long)(offE + sl) * IDIM;
      const float g0 = accg0[i], u0 = accu0[i];
      const float g1 = accg1[i], u1 = accu1[i];
      const float h0 = (g0 / (1.0f + __expf(-g0))) * u0;
      const float h1 = (g1 / (1.0f + __expf(-g1))) * u1;
      hb[hrow + n0 + colb + nl]      = f2bf(h0);
      hb[hrow + n0 + colb + 16 + nl] = f2bf(h1);
    }
  }
}

// ---------------------------------------------------------------------------
// Kernel 4: grouped GEMM2  out[t] += w_route * (h * w2)   (atomic combine)
// grid: (HDIM/BN, T/BM, NEXP), block 256, double-buffered async pipeline
// ---------------------------------------------------------------------------
__global__ __launch_bounds__(256) void moe_ffn2(
    const unsigned short* __restrict__ hb, const unsigned short* __restrict__ w2t,
    const int* __restrict__ list_tok, const float* __restrict__ list_w,
    const int* __restrict__ counts, const int* __restrict__ offsets,
    float* __restrict__ out) {
  const int e   = blockIdx.z;
  const int mt  = blockIdx.y;
  const int n0  = blockIdx.x * BN;
  const int cnt = counts[e];
  if (mt * BM >= cnt) return;
  const int offE = offsets[e];

  __shared__ __align__(16) unsigned short la[2][BM * KTP];
  __shared__ __align__(16) unsigned short lb[2][BN * KTP];

  const int tid  = threadIdx.x;
  const int lane = tid & 31;
  const int wave = tid >> 5;
  const int rowt = (wave >> 2) * 16;
  const int colb = (wave & 3) * 32;

  const int arow  = tid >> 3;
  const int acg   = (tid & 7) * 4;
  const int aslot = mt * BM + arow;
  const unsigned short* asrc =
      hb + (unsigned long long)(offE + ((aslot < cnt) ? aslot : 0)) * IDIM + acg;

  const int br0 = tid >> 2;
  const int bck = (tid & 3) * 8;
  const unsigned short* bsrc =
      w2t + (unsigned long long)e * IDIM * HDIM +
      (unsigned long long)(n0 + br0) * IDIM + bck;

  unsigned aoff[2], boff[2];
#pragma unroll
  for (int b = 0; b < 2; ++b) {
    aoff[b] = lds_off32(&la[b][arow * KTP + acg]);
    boff[b] = lds_off32(&lb[b][br0 * KTP + bck]);
  }

  auto issue = [&](int it, int buf) {          // 3 async ops per wave
    const int kt = it * KT;
    async_ld_b64 (aoff[buf],                asrc + kt);
    async_ld_b128(boff[buf],                bsrc + kt);
    async_ld_b128(boff[buf] + 64 * KTP * 2, bsrc + kt + (unsigned long long)64 * IDIM);
  };

  f32x8 acc0 = {}, acc1 = {};

  issue(0, 0);
  for (int it = 0; it < NST2; ++it) {
    const int cur = it & 1;
    if (it + 1 < NST2) {
      issue(it + 1, cur ^ 1);
      if (it + 2 < NST2) __builtin_prefetch(bsrc + (it + 2) * KT, 0, 1);
      WAIT_ASYNC_LE(3);
    } else {
      WAIT_ASYNC_LE(0);
    }
    __syncthreads();

    const FragBF fa = load_frag(la[cur], rowt,      lane);
    const FragBF b0 = load_frag(lb[cur], colb,      lane);
    const FragBF b1 = load_frag(lb[cur], colb + 16, lane);
    acc0 = WMMA_BF16(fa.v, b0.v, acc0);
    acc1 = WMMA_BF16(fa.v, b1.v, acc1);
    __syncthreads();
  }

  const int nl = lane & 15;
  const int mh = (lane >> 4) * 8;
#pragma unroll
  for (int i = 0; i < 8; ++i) {
    const int sl = mt * BM + rowt + mh + i;
    if (sl < cnt) {
      const int   tok = list_tok[e * T_TOK + sl];
      const float wgt = list_w[e * T_TOK + sl];
      float* orow = out + (unsigned long long)tok * HDIM;
      atomicAdd(&orow[n0 + colb + nl],      acc0[i] * wgt);
      atomicAdd(&orow[n0 + colb + 16 + nl], acc1[i] * wgt);
    }
  }
}

// ---------------------------------------------------------------------------
extern "C" void kernel_launch(void* const* d_in, const int* in_sizes, int n_in,
                              void* d_out, int out_size, void* d_ws, size_t ws_size,
                              hipStream_t stream) {
  (void)in_sizes; (void)n_in; (void)out_size; (void)ws_size;
  const float* x   = (const float*)d_in[0];   // [T, H]
  const float* gw  = (const float*)d_in[1];   // [E, H]
  const float* w1  = (const float*)d_in[2];   // [E, H, I]
  const float* w3  = (const float*)d_in[3];   // [E, H, I]
  const float* w2  = (const float*)d_in[4];   // [E, I, H]
  float*       out = (float*)d_out;           // [T, H]

  // workspace carve-up
  char* ws = (char*)d_ws;
  int*            counts   = (int*)ws;                          // 8
  int*            offsets  = counts + NEXP;                     // 8
  int*            list_tok = offsets + NEXP;                    // E*T
  float*          list_w   = (float*)(list_tok + NEXP * T_TOK); // E*T
  unsigned short* xb       = (unsigned short*)(list_w + NEXP * T_TOK);  // T*H
  unsigned short* hb       = xb  + (size_t)T_TOK * HDIM;        // 2T*I
  unsigned short* w1t      = hb  + (size_t)2 * T_TOK * IDIM;    // E*I*H  [E][I][H]
  unsigned short* w3t      = w1t + (size_t)NEXP * HDIM * IDIM;  // E*I*H
  unsigned short* w2t      = w3t + (size_t)NEXP * HDIM * IDIM;  // E*H*I  [E][H][I]

  moe_prep  <<<2048, 256, 0, stream>>>(x, out, xb, counts);
  moe_tcvt  <<<dim3(IDIM / 32, HDIM / 32, NEXP), 256, 0, stream>>>(w1, w1t, HDIM, IDIM);
  moe_tcvt  <<<dim3(IDIM / 32, HDIM / 32, NEXP), 256, 0, stream>>>(w3, w3t, HDIM, IDIM);
  moe_tcvt  <<<dim3(HDIM / 32, IDIM / 32, NEXP), 256, 0, stream>>>(w2, w2t, IDIM, HDIM);
  moe_router<<<T_TOK / 256, 256, 0, stream>>>(x, gw, counts, list_tok, list_w);
  moe_prefix<<<1, 32, 0, stream>>>(counts, offsets);
  moe_ffn1  <<<dim3(IDIM / BN, T_TOK / BM, NEXP), 256, 0, stream>>>(
      xb, w1t, w3t, list_tok, counts, offsets, hb);
  moe_ffn2  <<<dim3(HDIM / BN, T_TOK / BM, NEXP), 256, 0, stream>>>(
      hb, w2t, list_tok, list_w, counts, offsets, out);
}
